// MultiHead_Gated_Attention_68161130988102
// MI455X (gfx1250) — matrix operand order repaired and checked by reading, not compiled, not run on hardware
//
#include <hip/hip_runtime.h>

// ---------------------------------------------------------------------------
// MultiHead gated attention, CDNA5 (gfx1250, wave32, WMMA bf16 16x16x32)
// B=4, L=1024, SIZE=1024, BACK=512, H=16, D=64
// Pipeline: 4x projection GEMM (f32 -> bf16 heads) -> fused gated flash
// attention (never materializes LxL) -> output GEMM (bf16 -> f32).
// All tile staging goes through LDS; uses gfx1250 async global->LDS copies
// (ASYNCcnt) when the toolchain declares the builtins, else sync fallback.
// ---------------------------------------------------------------------------

typedef __attribute__((ext_vector_type(16))) __bf16 v16bf;
typedef __attribute__((ext_vector_type(8)))  float  v8f;

#define B_    4
#define L_    1024
#define SIZE_ 1024
#define BACK_ 512
#define H_    16
#define D_    64
#define M_    (B_ * L_)          // 4096 token rows

// ---- gfx1250 async global->LDS copy (guarded) ------------------------------
#if __has_builtin(__builtin_amdgcn_global_load_async_to_lds_b128) && \
    __has_builtin(__builtin_amdgcn_s_wait_asynccnt)
#define ASYNC_LDS 1
typedef int v4i_ __attribute__((vector_size(16)));
typedef __attribute__((address_space(1))) v4i_* gp16_t;   // global int4*
typedef __attribute__((address_space(3))) v4i_* lp16_t;   // LDS int4*
#else
#define ASYNC_LDS 0
#endif

struct alignas(16) Chunk16 { int x[4]; };

static __device__ __forceinline__ void cp16(void* lds_dst, const void* gsrc) {
#if ASYNC_LDS
  __builtin_amdgcn_global_load_async_to_lds_b128((gp16_t)gsrc, (lp16_t)lds_dst, 0, 0);
#else
  *(Chunk16*)lds_dst = *(const Chunk16*)gsrc;
#endif
}

static __device__ __forceinline__ void wait_async() {
#if ASYNC_LDS
  __builtin_amdgcn_s_wait_asynccnt(0);
#endif
}

static __device__ __forceinline__ int laneid() { return (int)(threadIdx.x & 31u); }

static __device__ __forceinline__ v8f wmma_bf16(v16bf a, v16bf b, v8f c) {
  // D = A(16x32) * B(32x16) + C(16x16 f32), wave32
  return __builtin_amdgcn_wmma_f32_16x16x32_bf16(false, a, false, b, (short)0, c, false, false);
}

// ---- Operand loaders -------------------------------------------------------
// A operand (16x32, M x K): lane m = lane&15; lanes 0-15 hold K 0..15,
// lanes 16-31 hold K 16..31. src row-major [rows][ld].
template <typename T>
static __device__ __forceinline__ v16bf load_a(const T* src, int ld, int row0, int k0) {
  const int l  = laneid();
  const int lo = (l >> 4) << 4;                       // 0 or 16
  const T* p = src + (size_t)(row0 + (l & 15)) * ld + k0;
  v16bf a;
#pragma unroll
  for (int i = 0; i < 8; ++i) a[i]     = (__bf16)(float)p[lo + i];
#pragma unroll
  for (int i = 0; i < 8; ++i) a[8 + i] = (__bf16)(float)p[lo + 8 + i];
  return a;
}

// B operand (32x16, K x N) where B = X^T of row-major X ([n-rows][ld]):
// column n = row (row0+n) of X; lanes 0-15 hold K 0..15, lanes 16-31 K 16..31.
static __device__ __forceinline__ v16bf load_bT(const __bf16* x, int ld, int row0, int k0) {
  const int l    = laneid();
  const int lo16 = (l >> 4) << 4;                     // 0 or 16
  const __bf16* p = x + (size_t)(row0 + (l & 15)) * ld + k0 + lo16;
  v16bf b;
#pragma unroll
  for (int i = 0; i < 16; ++i) b[i] = p[i];
  return b;
}

// B operand from a [K][ldl] (row-major K x N) tile, converting to bf16.
template <typename T>
static __device__ __forceinline__ v16bf load_b_kn(const T* w, int ldl, int n0) {
  const int l    = laneid();
  const int n    = n0 + (l & 15);
  const int lo16 = (l >> 4) << 4;
  v16bf b;
#pragma unroll
  for (int i = 0; i < 16; ++i) b[i] = (__bf16)(float)w[(size_t)(lo16 + i) * ldl + n];
  return b;
}

// ---- wave32 half-row reductions (16 lanes sharing a C-tile row) ------------
static __device__ __forceinline__ float rmax16(float v) {
#pragma unroll
  for (int m = 1; m <= 8; m <<= 1) v = fmaxf(v, __shfl_xor(v, m, 32));
  return v;
}
static __device__ __forceinline__ float rsum16(float v) {
#pragma unroll
  for (int m = 1; m <= 8; m <<= 1) v += __shfl_xor(v, m, 32);
  return v;
}

// ---------------------------------------------------------------------------
// GEMM: C[M,N] = A[M,Kd] @ W[Kd,N] + bias  (bf16 WMMA, f32 accumulate)
// Block: 128 threads / 4 waves; block tile 128(M) x 64(N); wave tile 32x64.
// A (128x32) and W (32x64) slices staged in LDS via async copies.
// MODE 0: bf16 -> per-head layout  dst[((b*H+h)*L + l)*D + d]
// MODE 1: bf16 -> transposed head  dst[((b*H+h)*D + d)*L + l]   (for V)
// MODE 2: f32  -> row-major        dst[r*N + c]                 (final out)
// ---------------------------------------------------------------------------
template <typename AT, int MODE>
__global__ __launch_bounds__(128)
void gemm_wmma(const AT* __restrict__ A, const float* __restrict__ W,
               const float* __restrict__ bias, void* __restrict__ outp,
               int Kd, int N, float scale) {
  constexpr int APAD = (sizeof(AT) == 4) ? 36 : 40;   // 16B-aligned padded rows
  __shared__ alignas(16) AT    Atile[128][APAD];
  __shared__ alignas(16) float Wtile[32][68];

  const int tid   = (int)threadIdx.x;
  const int lane  = tid & 31;
  const int wave  = tid >> 5;
  const int rows0 = (int)blockIdx.y * 128;
  const int col0  = (int)blockIdx.x * 64;
  const int wrow  = wave * 32;                        // wave's local row base

  v8f acc[2][4];
#pragma unroll
  for (int s = 0; s < 2; ++s)
#pragma unroll
    for (int t = 0; t < 4; ++t) acc[s][t] = (v8f)0.0f;

  constexpr int EPC = 16 / (int)sizeof(AT);           // elems per 16B chunk
  constexpr int CPR = 32 / EPC;                       // chunks per A row

  for (int k0 = 0; k0 < Kd; k0 += 32) {
    __syncthreads();
    // stage A[rows0..+127][k0..k0+31]
    for (int c = tid; c < 128 * CPR; c += 128) {
      const int r = c / CPR, s = c % CPR;
      cp16(&Atile[r][s * EPC], A + (size_t)(rows0 + r) * Kd + k0 + s * EPC);
    }
    // stage W[k0..k0+31][col0..col0+63]
    for (int c = tid; c < 512; c += 128) {
      const int r = c >> 4, s = c & 15;
      cp16(&Wtile[r][s * 4], W + (size_t)(k0 + r) * N + col0 + s * 4);
    }
    wait_async();
    __syncthreads();

    if (k0 + 32 < Kd) {                               // prefetch next K-slice
      __builtin_prefetch(W + (size_t)(k0 + 32) * N + col0 + (tid & 63), 0, 1);
      __builtin_prefetch(A + (size_t)(rows0 + tid) * Kd + k0 + 32, 0, 1);
    }

    v16bf b[4];
#pragma unroll
    for (int t = 0; t < 4; ++t) b[t] = load_b_kn<float>(&Wtile[0][0], 68, t * 16);
#pragma unroll
    for (int s = 0; s < 2; ++s) {
      const v16bf a = load_a<AT>(&Atile[0][0], APAD, wrow + s * 16, 0);
#pragma unroll
      for (int t = 0; t < 4; ++t) acc[s][t] = wmma_bf16(a, b[t], acc[s][t]);
    }
  }

  const int half8 = (lane >> 4) << 3;
#pragma unroll
  for (int s = 0; s < 2; ++s) {
#pragma unroll
    for (int t = 0; t < 4; ++t) {
#pragma unroll
      for (int j = 0; j < 8; ++j) {
        const int r = rows0 + wrow + s * 16 + j + half8;
        const int c = col0 + t * 16 + (lane & 15);
        const float v = (acc[s][t][j] + bias[c]) * scale;
        if (MODE == 2) {
          ((float*)outp)[(size_t)r * N + c] = v;
        } else {
          const int bb = r >> 10, l = r & (L_ - 1);
          const int h  = c >> 6,  d = c & (D_ - 1);
          __bf16* o = (__bf16*)outp;
          if (MODE == 0)
            o[(((size_t)bb * H_ + h) * L_ + l) * D_ + d] = (__bf16)v;
          else
            o[(((size_t)bb * H_ + h) * D_ + d) * L_ + l] = (__bf16)v;
        }
      }
    }
  }
}

// ---------------------------------------------------------------------------
// Fused gated attention (flash style):
//   S = Qh Kh^T ; G = sigmoid(Qh Gh^T) ; ctx = (softmax(S) * G) Vh
// Online stats: m (row max), lsum (pure exp sum), acc = sum exp*gate*V.
// Block: 4 waves, same (b,h); waves own 16-query tiles; K/G/V key-blocks
// (32 keys) staged once per block in LDS via async copies. 12 WMMA / iter.
// Qh/Kh/Gh: [B,H,L,D] bf16.  Vt: [B,H,D,L] bf16.  ctx out: [M_, SIZE_] bf16.
// ---------------------------------------------------------------------------
__global__ __launch_bounds__(128)
void attn_kernel(const __bf16* __restrict__ Qh, const __bf16* __restrict__ Kh,
                 const __bf16* __restrict__ Gh, const __bf16* __restrict__ Vt,
                 __bf16* __restrict__ ctx) {
  __shared__ alignas(16) __bf16 Ks[32][72];           // keys x d
  __shared__ alignas(16) __bf16 Gs[32][72];           // keys x d
  __shared__ alignas(16) __bf16 Vs[64][40];           // d x keys
  __shared__ alignas(16) __bf16 plds[4][16][40];      // per-wave gated-P 16x32

  const int tid  = (int)threadIdx.x;
  const int lane = tid & 31;
  const int wave = tid >> 5;
  const int bh   = (int)blockIdx.y;                   // 0..63
  const int b    = bh >> 4, h = bh & 15;
  const int q0   = ((int)blockIdx.x * 4 + wave) * 16;

  const __bf16* Q = Qh + (size_t)bh * L_ * D_;
  const __bf16* K = Kh + (size_t)bh * L_ * D_;
  const __bf16* G = Gh + (size_t)bh * L_ * D_;
  const __bf16* V = Vt + (size_t)bh * D_ * L_;

  // Q A-operands resident for the whole key loop (D=64 -> two K=32 chunks)
  const v16bf qa0 = load_a<__bf16>(Q, D_, q0, 0);
  const v16bf qa1 = load_a<__bf16>(Q, D_, q0, 32);

  v8f acc[4];
#pragma unroll
  for (int t = 0; t < 4; ++t) acc[t] = (v8f)0.0f;
  float m[8], lsum[8];
#pragma unroll
  for (int j = 0; j < 8; ++j) { m[j] = -1.0e30f; lsum[j] = 0.0f; }

  const int half8 = (lane >> 4) << 3;
  __bf16* pw = &plds[wave][0][0];

  for (int kb = 0; kb < L_; kb += 32) {
    __syncthreads();
    // stage K/G blocks: 32 rows x 64 bf16 (8 chunks/row)
    for (int c = tid; c < 256; c += 128) {
      const int r = c >> 3, s = c & 7;
      cp16(&Ks[r][s * 8], K + (size_t)(kb + r) * D_ + s * 8);
    }
    for (int c = tid; c < 256; c += 128) {
      const int r = c >> 3, s = c & 7;
      cp16(&Gs[r][s * 8], G + (size_t)(kb + r) * D_ + s * 8);
    }
    // stage V block: 64 rows x 32 bf16 (4 chunks/row)
    for (int c = tid; c < 256; c += 128) {
      const int r = c >> 2, s = c & 3;
      cp16(&Vs[r][s * 8], V + (size_t)r * L_ + kb + s * 8);
    }
    wait_async();
    __syncthreads();

    if (kb + 32 < L_) {                               // prefetch next key block
      __builtin_prefetch(K + (size_t)(kb + 32) * D_ + tid * 16, 0, 1);
      __builtin_prefetch(G + (size_t)(kb + 32) * D_ + tid * 16, 0, 1);
      __builtin_prefetch(V + (size_t)(tid >> 1) * L_ + kb + 32, 0, 1);
    }

    // ---- scores / gate logits: 16q x 32k, reduce over D=64 ----
    v8f s0 = (v8f)0.0f, s1 = (v8f)0.0f, g0 = (v8f)0.0f, g1 = (v8f)0.0f;
    s0 = wmma_bf16(qa0, load_bT(&Ks[0][0], 72, 0,  0),  s0);
    s0 = wmma_bf16(qa1, load_bT(&Ks[0][0], 72, 0,  32), s0);
    s1 = wmma_bf16(qa0, load_bT(&Ks[0][0], 72, 16, 0),  s1);
    s1 = wmma_bf16(qa1, load_bT(&Ks[0][0], 72, 16, 32), s1);
    g0 = wmma_bf16(qa0, load_bT(&Gs[0][0], 72, 0,  0),  g0);
    g0 = wmma_bf16(qa1, load_bT(&Gs[0][0], 72, 0,  32), g0);
    g1 = wmma_bf16(qa0, load_bT(&Gs[0][0], 72, 16, 0),  g1);
    g1 = wmma_bf16(qa1, load_bT(&Gs[0][0], 72, 16, 32), g1);

    // ---- online softmax update + gating, per C-tile row ----
#pragma unroll
    for (int j = 0; j < 8; ++j) {
      const float mx = rmax16(fmaxf(s0[j], s1[j]));
      const float mn = fmaxf(m[j], mx);
      const float sc = __expf(m[j] - mn);
      const float e0 = __expf(s0[j] - mn);
      const float e1 = __expf(s1[j] - mn);
      lsum[j] = lsum[j] * sc + rsum16(e0 + e1);       // denom: pure softmax sum
      m[j] = mn;
      acc[0][j] *= sc; acc[1][j] *= sc; acc[2][j] *= sc; acc[3][j] *= sc;
      const float p0 = e0 / (1.0f + __expf(-g0[j]));  // exp * sigmoid(gate)
      const float p1 = e1 / (1.0f + __expf(-g1[j]));
      // C-layout -> LDS (row q, col key) so it can be re-read as an A operand
      pw[(size_t)(j + half8) * 40 + (lane & 15)]      = (__bf16)p0;
      pw[(size_t)(j + half8) * 40 + 16 + (lane & 15)] = (__bf16)p1;
    }

    // ---- ctx += P(16x32) @ Vblk(32x64) ----
    const v16bf pa = load_a<__bf16>(pw, 40, 0, 0);
#pragma unroll
    for (int t = 0; t < 4; ++t)
      acc[t] = wmma_bf16(pa, load_bT(&Vs[0][0], 40, t * 16, 0), acc[t]);
  }

  // ---- normalize and write ctx as bf16 row-major [M_, SIZE_] ----
#pragma unroll
  for (int t = 0; t < 4; ++t) {
#pragma unroll
    for (int j = 0; j < 8; ++j) {
      const float v = acc[t][j] / lsum[j];
      const int r = b * L_ + q0 + j + half8;
      const int c = h * D_ + t * 16 + (lane & 15);
      ctx[(size_t)r * SIZE_ + c] = (__bf16)v;
    }
  }
}

// ---------------------------------------------------------------------------
extern "C" void kernel_launch(void* const* d_in, const int* in_sizes, int n_in,
                              void* d_out, int out_size, void* d_ws, size_t ws_size,
                              hipStream_t stream) {
  (void)in_sizes; (void)n_in; (void)out_size; (void)ws_size;
  const float* k   = (const float*)d_in[0];
  const float* v   = (const float*)d_in[1];
  const float* q   = (const float*)d_in[2];
  const float* bg  = (const float*)d_in[3];
  const float* Wk  = (const float*)d_in[4];
  const float* bk  = (const float*)d_in[5];
  const float* Wv  = (const float*)d_in[6];
  const float* bv  = (const float*)d_in[7];
  const float* Wq  = (const float*)d_in[8];
  const float* bq  = (const float*)d_in[9];
  const float* Wb  = (const float*)d_in[10];
  const float* bb  = (const float*)d_in[11];
  const float* Wo  = (const float*)d_in[12];
  const float* bo  = (const float*)d_in[13];

  const size_t HEAD_BYTES = (size_t)B_ * H_ * L_ * D_ * sizeof(__bf16); // 8 MiB
  char* ws = (char*)d_ws;
  __bf16* Qh  = (__bf16*)(ws + 0 * HEAD_BYTES);
  __bf16* Kh  = (__bf16*)(ws + 1 * HEAD_BYTES);
  __bf16* Gh  = (__bf16*)(ws + 2 * HEAD_BYTES);
  __bf16* Vt  = (__bf16*)(ws + 3 * HEAD_BYTES);
  __bf16* ctx = (__bf16*)(ws + 4 * HEAD_BYTES);

  const dim3 blk(128, 1, 1);
  const dim3 gproj(SIZE_ / 64, M_ / 128, 1);          // (16, 32)

  // Projections -> bf16 head layouts (Q fused with 1/sqrt(D))
  gemm_wmma<float, 0><<<gproj, blk, 0, stream>>>(k,  Wk, bk, (void*)Kh, SIZE_, SIZE_, 1.0f);
  gemm_wmma<float, 1><<<gproj, blk, 0, stream>>>(v,  Wv, bv, (void*)Vt, SIZE_, SIZE_, 1.0f);
  gemm_wmma<float, 0><<<gproj, blk, 0, stream>>>(q,  Wq, bq, (void*)Qh, SIZE_, SIZE_, 0.125f);
  gemm_wmma<float, 0><<<gproj, blk, 0, stream>>>(bg, Wb, bb, (void*)Gh, BACK_, SIZE_, 1.0f);

  // Fused gated flash attention
  const dim3 gattn(L_ / 64, B_ * H_, 1);              // (16, 64)
  attn_kernel<<<gattn, blk, 0, stream>>>(Qh, Kh, Gh, Vt, ctx);

  // Output projection -> f32
  gemm_wmma<__bf16, 2><<<gproj, blk, 0, stream>>>(ctx, Wo, bo, d_out, SIZE_, SIZE_, 1.0f);
}
